// ViT_7335804141849
// MI455X (gfx1250) — 8-bit matrix operand order repaired and checked by reading, compiled, not run on hardware
//
#include <hip/hip_runtime.h>
#include <hip/hip_bf16.h>
#include <stdint.h>

typedef __attribute__((ext_vector_type(8))) int v8i;

#define NHEADS   12
#define DIMH     64
#define DEPTH    12
#define DMODEL   768
#define NTOK     197
#define NPATCH   196
#define BATCH    32
#define MTOK     (BATCH * NTOK)
#define MPATCH   (BATCH * NPATCH)
#define FFDIM    3072
#define NCLS     1000

__global__ void patchify_ln_kernel(const float* __restrict__ img,
                                   const float* __restrict__ g,
                                   const float* __restrict__ be,
                                   float* __restrict__ Y) {
    __shared__ float buf[DMODEL];
    __shared__ float red[256];
    int row = blockIdx.x;
    int b   = row / NPATCH;
    int pp  = row - b * NPATCH;
    int ph  = pp / 14, pw = pp - ph * 14;
    int tid = threadIdx.x;

    float psum = 0.f, psq = 0.f;
    for (int i = tid; i < DMODEL; i += 256) {
        int p = i / 48;
        int r = i - p * 48;
        int q = r / 3;
        int c = r - q * 3;
        float v = img[(((size_t)b * 3 + c) * 224 + (ph * 16 + p)) * 224 + (pw * 16 + q)];
        buf[i] = v;
        psum += v; psq += v * v;
    }
    red[tid] = psum; __syncthreads();
    for (int s = 128; s > 0; s >>= 1) { if (tid < s) red[tid] += red[tid + s]; __syncthreads(); }
    float mean = red[0] * (1.f / DMODEL); __syncthreads();
    red[tid] = psq; __syncthreads();
    for (int s = 128; s > 0; s >>= 1) { if (tid < s) red[tid] += red[tid + s]; __syncthreads(); }
    float var  = red[0] * (1.f / DMODEL) - mean * mean;
    float rstd = rsqrtf(var + 1e-5f);
    for (int i = tid; i < DMODEL; i += 256)
        Y[(size_t)row * DMODEL + i] = (buf[i] - mean) * rstd * g[i] + be[i];
}

__global__ void ln_kernel(const float* __restrict__ X, long in_stride,
                          const float* __restrict__ g, const float* __restrict__ be,
                          float* __restrict__ Y) {
    __shared__ float buf[DMODEL];
    __shared__ float red[256];
    int row = blockIdx.x;
    int tid = threadIdx.x;
    const float* xp = X + (size_t)row * in_stride;

    float psum = 0.f, psq = 0.f;
    for (int i = tid; i < DMODEL; i += 256) {
        float v = xp[i];
        buf[i] = v; psum += v; psq += v * v;
    }
    red[tid] = psum; __syncthreads();
    for (int s = 128; s > 0; s >>= 1) { if (tid < s) red[tid] += red[tid + s]; __syncthreads(); }
    float mean = red[0] * (1.f / DMODEL); __syncthreads();
    red[tid] = psq; __syncthreads();
    for (int s = 128; s > 0; s >>= 1) { if (tid < s) red[tid] += red[tid + s]; __syncthreads(); }
    float var  = red[0] * (1.f / DMODEL) - mean * mean;
    float rstd = rsqrtf(var + 1e-5f);
    for (int i = tid; i < DMODEL; i += 256)
        Y[(size_t)row * DMODEL + i] = (buf[i] - mean) * rstd * g[i] + be[i];
}

__global__ void assemble_kernel(const float* __restrict__ emb,
                                const float* __restrict__ g, const float* __restrict__ be,
                                const float* __restrict__ cls, const float* __restrict__ pos,
                                float* __restrict__ X) {
    __shared__ float buf[DMODEL];
    __shared__ float red[256];
    int row = blockIdx.x;
    int b = row / NTOK, t = row - b * NTOK;
    int tid = threadIdx.x;
    if (t == 0) {
        for (int i = tid; i < DMODEL; i += 256)
            X[(size_t)row * DMODEL + i] = cls[i] + pos[i];
        return;
    }
    const float* xp = emb + ((size_t)b * NPATCH + (t - 1)) * DMODEL;
    float psum = 0.f, psq = 0.f;
    for (int i = tid; i < DMODEL; i += 256) {
        float v = xp[i]; buf[i] = v; psum += v; psq += v * v;
    }
    red[tid] = psum; __syncthreads();
    for (int s = 128; s > 0; s >>= 1) { if (tid < s) red[tid] += red[tid + s]; __syncthreads(); }
    float mean = red[0] * (1.f / DMODEL); __syncthreads();
    red[tid] = psq; __syncthreads();
    for (int s = 128; s > 0; s >>= 1) { if (tid < s) red[tid] += red[tid + s]; __syncthreads(); }
    float var  = red[0] * (1.f / DMODEL) - mean * mean;
    float rstd = rsqrtf(var + 1e-5f);
    for (int i = tid; i < DMODEL; i += 256)
        X[(size_t)row * DMODEL + i] =
            (buf[i] - mean) * rstd * g[i] + be[i] + pos[(size_t)t * DMODEL + i];
}

__global__ void rowquant_kernel(const float* __restrict__ X, long in_stride, int K,
                                int8_t* __restrict__ Q, float* __restrict__ S) {
    __shared__ float red[256];
    int row = blockIdx.x;
    int tid = threadIdx.x;
    const float* xp = X + (size_t)row * in_stride;
    float m = 0.f;
    for (int i = tid; i < K; i += 256) m = fmaxf(m, fabsf(xp[i]));
    red[tid] = m; __syncthreads();
    for (int s = 128; s > 0; s >>= 1) { if (tid < s) red[tid] = fmaxf(red[tid], red[tid + s]); __syncthreads(); }
    float scale = fmaxf(red[0], 1e-8f) * (1.f / 127.f);
    float inv   = 1.f / scale;
    for (int i = tid; i < K; i += 256) {
        float q = rintf(xp[i] * inv);
        q = fminf(fmaxf(q, -128.f), 127.f);
        Q[(size_t)row * K + i] = (int8_t)q;
    }
    if (tid == 0) S[row] = scale;
}

__global__ void gemm_i8_kernel(const int8_t* __restrict__ A,  const float* __restrict__ As,
                               const int8_t* __restrict__ Bq, const float* __restrict__ Bs,
                               const float* __restrict__ bias, const float* __restrict__ res,
                               float* __restrict__ C, int M, int N, int K, int act) {
    int lane = threadIdx.x;
    int tm   = blockIdx.y * 4 + threadIdx.y;
    int tn4  = blockIdx.x;
    if (tm * 16 >= M) return;

    int arow0 = tm * 16 + (lane & 15);
    int arow  = arow0 < M ? arow0 : M - 1;
    int koffA = (lane >= 16) ? 16 : 0;
    int koffB = (lane >= 16) ? 16 : 0;

    const int8_t* arowp = A + (size_t)arow * K;
    const int8_t* browp[4];
    int colIn[4], colC[4];
    #pragma unroll
    for (int j = 0; j < 4; ++j) {
        colIn[j] = tn4 * 64 + j * 16 + (lane & 15);
        colC[j]  = colIn[j] < N ? colIn[j] : N - 1;
        browp[j] = Bq + (size_t)colC[j] * K;
    }

    v8i acc0 = {0,0,0,0,0,0,0,0};
    v8i acc1 = {0,0,0,0,0,0,0,0};
    v8i acc2 = {0,0,0,0,0,0,0,0};
    v8i acc3 = {0,0,0,0,0,0,0,0};

    for (int kk = 0; kk < K; kk += 64) {
        const int* ap = (const int*)(arowp + kk + koffA);
        v8i av;
        av[0] = ap[0];  av[1] = ap[1];
        av[2] = ap[2];  av[3] = ap[3];
        av[4] = ap[8];  av[5] = ap[9];
        av[6] = ap[10]; av[7] = ap[11];

        v8i bv[4];
        #pragma unroll
        for (int j = 0; j < 4; ++j) {
            const int* bp = (const int*)(browp[j] + kk + koffB);
            bv[j][0] = bp[0];  bv[j][1] = bp[1];  bv[j][2] = bp[2];  bv[j][3] = bp[3];
            bv[j][4] = bp[8];  bv[j][5] = bp[9];  bv[j][6] = bp[10]; bv[j][7] = bp[11];
        }
        acc0 = __builtin_amdgcn_wmma_i32_16x16x64_iu8(true, av, true, bv[0], acc0, false, false);
        acc1 = __builtin_amdgcn_wmma_i32_16x16x64_iu8(true, av, true, bv[1], acc1, false, false);
        acc2 = __builtin_amdgcn_wmma_i32_16x16x64_iu8(true, av, true, bv[2], acc2, false, false);
        acc3 = __builtin_amdgcn_wmma_i32_16x16x64_iu8(true, av, true, bv[3], acc3, false, false);
    }

    int rbase = tm * 16 + ((lane >= 16) ? 8 : 0);
    float asr[8];
    #pragma unroll
    for (int r = 0; r < 8; ++r) {
        int row = rbase + r;
        asr[r] = As[row < M ? row : M - 1];
    }
    v8i accs[4] = {acc0, acc1, acc2, acc3};
    #pragma unroll
    for (int j = 0; j < 4; ++j) {
        if (colIn[j] >= N) continue;
        float wscale = Bs[colC[j]];
        float bval   = bias ? bias[colC[j]] : 0.f;
        #pragma unroll
        for (int r = 0; r < 8; ++r) {
            int row = rbase + r;
            if (row >= M) continue;
            float v = (float)accs[j][r] * asr[r] * wscale + bval;
            if (act == 1) v = 0.5f * v * (1.f + erff(v * 0.70710678118654752f));
            if (res) v += res[(size_t)row * N + colIn[j]];
            C[(size_t)row * N + colIn[j]] = v;
        }
    }
}

__global__ void attention_kernel(const float* __restrict__ qkv, float* __restrict__ out) {
    __shared__ float probs[224];
    __shared__ float qs[DIMH];
    __shared__ float red[64];
    int t = blockIdx.x, h = blockIdx.y, b = blockIdx.z;
    int tid = threadIdx.x;

    qs[tid] = qkv[((size_t)b * NTOK + t) * (3 * DMODEL) + h * DIMH + tid];
    __syncthreads();

    float dloc[4];
    float pm = -1e30f;
    for (int it = 0; it < 4; ++it) {
        int j = tid + it * 64;
        float d = -1e30f;
        if (j < NTOK) {
            const float* kp = qkv + ((size_t)b * NTOK + j) * (3 * DMODEL) + DMODEL + h * DIMH;
            float s = 0.f;
            #pragma unroll
            for (int dd = 0; dd < DIMH; ++dd) s += qs[dd] * kp[dd];
            d = s * 0.125f;
        }
        dloc[it] = d;
        pm = fmaxf(pm, d);
    }
    red[tid] = pm; __syncthreads();
    for (int s = 32; s > 0; s >>= 1) { if (tid < s) red[tid] = fmaxf(red[tid], red[tid + s]); __syncthreads(); }
    float mx = red[0]; __syncthreads();

    float psum = 0.f;
    for (int it = 0; it < 4; ++it) {
        int j = tid + it * 64;
        if (j < 224) {
            float e = (j < NTOK) ? expf(dloc[it] - mx) : 0.f;
            probs[j] = e; psum += e;
        }
    }
    red[tid] = psum; __syncthreads();
    for (int s = 32; s > 0; s >>= 1) { if (tid < s) red[tid] += red[tid + s]; __syncthreads(); }
    float inv = 1.f / red[0];

    float acc = 0.f;
    for (int j = 0; j < NTOK; ++j) {
        const float* vp = qkv + ((size_t)b * NTOK + j) * (3 * DMODEL) + 2 * DMODEL + h * DIMH;
        acc += probs[j] * vp[tid];
    }
    out[((size_t)b * NTOK + t) * DMODEL + h * DIMH + tid] = acc * inv;
}

extern "C" void kernel_launch(void* const* d_in, const int* in_sizes, int n_in,
                              void* d_out, int out_size, void* d_ws, size_t ws_size,
                              hipStream_t stream) {
    const float* img        = (const float*)d_in[0];
    const float* ln_patch_g = (const float*)d_in[1];
    const float* ln_patch_b = (const float*)d_in[2];
    const float* Wp         = (const float*)d_in[3];
    const float* bp         = (const float*)d_in[4];
    const float* ln_post_g  = (const float*)d_in[5];
    const float* ln_post_b  = (const float*)d_in[6];
    const float* cls_token  = (const float*)d_in[7];
    const float* pos_emb    = (const float*)d_in[8];
    const float* attn_ln_g  = (const float*)d_in[9];
    const float* attn_ln_b  = (const float*)d_in[10];
    const float* Wqkv       = (const float*)d_in[11];
    const float* Wo         = (const float*)d_in[12];
    const float* bo         = (const float*)d_in[13];
    const float* ff_ln_g    = (const float*)d_in[14];
    const float* ff_ln_b    = (const float*)d_in[15];
    const float* W1         = (const float*)d_in[16];
    const float* b1         = (const float*)d_in[17];
    const float* W2         = (const float*)d_in[18];
    const float* b2         = (const float*)d_in[19];
    const float* ln_f_g     = (const float*)d_in[20];
    const float* ln_f_b     = (const float*)d_in[21];
    const float* Wh         = (const float*)d_in[22];
    const float* bh         = (const float*)d_in[23];

    uint8_t* ws = (uint8_t*)d_ws;
    size_t off = 0;
    auto alloc = [&](size_t bytes) -> void* {
        off = (off + 255) & ~(size_t)255;
        void* p = ws + off;
        off += bytes;
        return p;
    };
    float*  X   = (float*) alloc((size_t)MTOK * DMODEL * 4);
    float*  Y   = (float*) alloc((size_t)MTOK * DMODEL * 4);
    float*  QKV = (float*) alloc((size_t)MTOK * 3 * DMODEL * 4);
    float*  FFB = (float*) alloc((size_t)MTOK * FFDIM * 4);
    int8_t* AQ  = (int8_t*)alloc((size_t)MTOK * FFDIM);
    float*  AS  = (float*) alloc((size_t)MTOK * 4);
    int8_t* WQ  = (int8_t*)alloc((size_t)FFDIM * DMODEL);
    float*  WS  = (float*) alloc((size_t)FFDIM * 4);
    (void)ws_size; (void)in_sizes; (void)n_in; (void)out_size;

    const dim3 gblk(32, 4);
    auto gemm = [&](const int8_t* A, const float* as, const int8_t* Bq, const float* bs,
                    const float* bias, const float* res, float* C,
                    int M, int N, int K, int act) {
        dim3 grid((N + 63) / 64, ((M + 15) / 16 + 3) / 4);
        hipLaunchKernelGGL(gemm_i8_kernel, grid, gblk, 0, stream,
                           A, as, Bq, bs, bias, res, C, M, N, K, act);
    };

    hipLaunchKernelGGL(patchify_ln_kernel, dim3(MPATCH), dim3(256), 0, stream,
                       img, ln_patch_g, ln_patch_b, Y);
    hipLaunchKernelGGL(rowquant_kernel, dim3(MPATCH), dim3(256), 0, stream,
                       Y, (long)DMODEL, DMODEL, AQ, AS);
    hipLaunchKernelGGL(rowquant_kernel, dim3(DMODEL), dim3(256), 0, stream,
                       Wp, (long)DMODEL, DMODEL, WQ, WS);
    gemm(AQ, AS, WQ, WS, bp, nullptr, QKV  , MPATCH, DMODEL, DMODEL, 0);
    hipLaunchKernelGGL(assemble_kernel, dim3(MTOK), dim3(256), 0, stream,
                       QKV, ln_post_g, ln_post_b, cls_token, pos_emb, X);

    for (int l = 0; l < DEPTH; ++l) {
        hipLaunchKernelGGL(ln_kernel, dim3(MTOK), dim3(256), 0, stream,
                           X, (long)DMODEL, attn_ln_g + (size_t)l * DMODEL,
                           attn_ln_b + (size_t)l * DMODEL, Y);
        hipLaunchKernelGGL(rowquant_kernel, dim3(MTOK), dim3(256), 0, stream,
                           Y, (long)DMODEL, DMODEL, AQ, AS);
        hipLaunchKernelGGL(rowquant_kernel, dim3(3 * DMODEL), dim3(256), 0, stream,
                           Wqkv + (size_t)l * 3 * DMODEL * DMODEL, (long)DMODEL, DMODEL, WQ, WS);
        gemm(AQ, AS, WQ, WS, nullptr, nullptr, QKV, MTOK, 3 * DMODEL, DMODEL, 0);
        hipLaunchKernelGGL(attention_kernel, dim3(NTOK, NHEADS, BATCH), dim3(64), 0, stream,
                           QKV, Y);
        hipLaunchKernelGGL(rowquant_kernel, dim3(MTOK), dim3(256), 0, stream,
                           Y, (long)DMODEL, DMODEL, AQ, AS);
        hipLaunchKernelGGL(rowquant_kernel, dim3(DMODEL), dim3(256), 0, stream,
                           Wo + (size_t)l * DMODEL * DMODEL, (long)DMODEL, DMODEL, WQ, WS);
        gemm(AQ, AS, WQ, WS, bo + (size_t)l * DMODEL, X  , X, MTOK, DMODEL, DMODEL, 0);

        hipLaunchKernelGGL(ln_kernel, dim3(MTOK), dim3(256), 0, stream,
                           X, (long)DMODEL, ff_ln_g + (size_t)l * DMODEL,
                           ff_ln_b + (size_t)l * DMODEL, Y);
        hipLaunchKernelGGL(rowquant_kernel, dim3(MTOK), dim3(256), 0, stream,
                           Y, (long)DMODEL, DMODEL, AQ, AS);
        hipLaunchKernelGGL(rowquant_kernel, dim3(FFDIM), dim3(256), 0, stream,
                           W1 + (size_t)l * FFDIM * DMODEL, (long)DMODEL, DMODEL, WQ, WS);
        gemm(AQ, AS, WQ, WS, b1 + (size_t)l * FFDIM, nullptr, FFB, MTOK, FFDIM, DMODEL, 1  );
        hipLaunchKernelGGL(rowquant_kernel, dim3(MTOK), dim3(256), 0, stream,
                           FFB, (long)FFDIM, FFDIM, AQ, AS);
        hipLaunchKernelGGL(rowquant_kernel, dim3(DMODEL), dim3(256), 0, stream,
                           W2 + (size_t)l * DMODEL * FFDIM, (long)FFDIM, FFDIM, WQ, WS);
        gemm(AQ, AS, WQ, WS, b2 + (size_t)l * DMODEL, X  , X, MTOK, DMODEL, FFDIM, 0);
    }

    hipLaunchKernelGGL(ln_kernel, dim3(BATCH), dim3(256), 0, stream,
                       X, (long)NTOK * DMODEL, ln_f_g, ln_f_b, Y);
    hipLaunchKernelGGL(rowquant_kernel, dim3(BATCH), dim3(256), 0, stream,
                       Y, (long)DMODEL, DMODEL, AQ, AS);
    hipLaunchKernelGGL(rowquant_kernel, dim3(NCLS), dim3(256), 0, stream,
                       Wh, (long)DMODEL, DMODEL, WQ, WS);
    gemm(AQ, AS, WQ, WS, bh, nullptr, (float*)d_out, BATCH, NCLS, DMODEL, 0);
}
